// LSTMmodel_36730560315621
// MI455X (gfx1250) — matrix operand order repaired and checked by reading, not compiled, not run on hardware
//
#include <hip/hip_runtime.h>

typedef __attribute__((ext_vector_type(16))) _Float16 v16h;
typedef __attribute__((ext_vector_type(8)))  float    v8f;

#define B_   64
#define LP_  256
#define LH_  256
#define E_   512
#define H_   1024
#define G4H_ 4096

// ---------------- utility kernels ----------------

__global__ void cast_f32_to_f16(const float* __restrict__ in, _Float16* __restrict__ out, int n) {
  int i = blockIdx.x * blockDim.x + threadIdx.x;
  if (i < n) out[i] = (_Float16)in[i];
}

__global__ void gather_embed(const float* __restrict__ emb, const int* __restrict__ idx,
                             _Float16* __restrict__ out, int total) {
  int i = blockIdx.x * blockDim.x + threadIdx.x;
  if (i >= total) return;
  int r = i >> 9;          // / E_
  int k = i & (E_ - 1);    // % E_
  out[i] = (_Float16)emb[(long)idx[r] * E_ + k];
}

__global__ void zero_state(float* __restrict__ h, float* __restrict__ c,
                           _Float16* __restrict__ h16, int n) {
  int i = blockIdx.x * blockDim.x + threadIdx.x;
  if (i < n) { h[i] = 0.f; c[i] = 0.f; h16[i] = (_Float16)0.f; }
}

// ---------------- WMMA GEMM, 32x32 register-blocked tile per wave --------------------------
// A row-major [M,K] f16, W row-major [N,K] f16 (i.e. B = W^T), C fp32 [M,N].
// C[m,n] = (beta ? C[m,n] : 0) + sum_k A[m,k]*W[n,k] + bias[n]
// Each wave: 2 M-subtiles x 2 N-subtiles -> 4 accumulators, 4 v_wmma per K-step of 32.
// A fragments reused across the 2 N-subtiles, B fragments across the 2 M-subtiles:
// 64 KFLOP per 4KB of operand traffic (16 FLOP/B) vs 8 FLOP/B for a flat 16x16 tile.

union Frag16 { v16h h; unsigned u[8]; };

__device__ __forceinline__ void load_afrag(Frag16& f, const _Float16* __restrict__ p, int hi) {
#pragma unroll
  for (int q = 0; q < 8; ++q) {
    int ka = 2 * q + hi * 16;     // A 16x32 f16 layout (contiguous K per lane half)
    f.u[q] = *reinterpret_cast<const unsigned*>(p + ka);
  }
}
__device__ __forceinline__ void load_bfrag(Frag16& f, const _Float16* __restrict__ p) {
#pragma unroll
  for (int q = 0; q < 8; ++q)                       // B 32x16 f16 layout: K pairs per VGPR
    f.u[q] = *reinterpret_cast<const unsigned*>(p + 2 * q);
}

__global__ void wmma_gemm_nt(const _Float16* __restrict__ A, int lda,
                             const _Float16* __restrict__ W, int ldb,
                             float* __restrict__ C, int ldc,
                             int Mt2, int Nt2, int K,
                             const float* __restrict__ bias, int beta) {
  int wave  = blockIdx.x * (blockDim.x >> 5) + (threadIdx.x >> 5);
  int tiles = Mt2 * Nt2;
  if (wave >= tiles) return;              // whole-wave exit: EXEC stays all-ones below
  int mt = wave / Nt2;                    // 32-row tile
  int nt = wave % Nt2;                    // 32-col tile
  int lane = threadIdx.x & 31;
  int hi   = (lane >> 4) & 1;
  int nn   = lane & 15;

  v8f acc00 = {}, acc01 = {}, acc10 = {}, acc11 = {};
  if (beta) {
#pragma unroll
    for (int v = 0; v < 8; ++v) {
      long m0 = (long)(mt * 32 + v + hi * 8) * ldc;
      long m1 = m0 + (long)16 * ldc;
      int  n0 = nt * 32 + nn;
      acc00[v] = C[m0 + n0];       acc01[v] = C[m0 + n0 + 16];
      acc10[v] = C[m1 + n0];       acc11[v] = C[m1 + n0 + 16];
    }
  }

  const _Float16* A0 = A + (long)(mt * 32 + nn) * lda;
  const _Float16* A1 = A0 + (long)16 * lda;
  const _Float16* W0 = W + (long)(nt * 32 + nn) * ldb + hi * 16;
  const _Float16* W1 = W0 + (long)16 * ldb;

  for (int k = 0; k < K; k += 32) {
    Frag16 a0, a1, b0, b1;
    __builtin_prefetch(W0 + k + 128, 0, 1);   // stream next weight chunk (global_prefetch_b8)
    __builtin_prefetch(W1 + k + 128, 0, 1);
    load_afrag(a0, A0 + k, hi);
    load_afrag(a1, A1 + k, hi);
    load_bfrag(b0, W0 + k);
    load_bfrag(b1, W1 + k);
    acc00 = __builtin_amdgcn_wmma_f32_16x16x32_f16(false, a0.h, false, b0.h, (short)0, acc00, false, false);
    acc01 = __builtin_amdgcn_wmma_f32_16x16x32_f16(false, a0.h, false, b1.h, (short)0, acc01, false, false);
    acc10 = __builtin_amdgcn_wmma_f32_16x16x32_f16(false, a1.h, false, b0.h, (short)0, acc10, false, false);
    acc11 = __builtin_amdgcn_wmma_f32_16x16x32_f16(false, a1.h, false, b1.h, (short)0, acc11, false, false);
  }

  float bv0 = bias ? bias[nt * 32 + nn]      : 0.f;
  float bv1 = bias ? bias[nt * 32 + nn + 16] : 0.f;
#pragma unroll
  for (int v = 0; v < 8; ++v) {
    long m0 = (long)(mt * 32 + v + hi * 8) * ldc;
    long m1 = m0 + (long)16 * ldc;
    int  n0 = nt * 32 + nn;
    C[m0 + n0]      = acc00[v] + bv0;
    C[m0 + n0 + 16] = acc01[v] + bv1;
    C[m1 + n0]      = acc10[v] + bv0;
    C[m1 + n0 + 16] = acc11[v] + bv1;
  }
}

// ---------------- LSTM pointwise: gates -> (h,c) with packed-sequence masking ----------------

__device__ __forceinline__ float sigm_(float x) { return 1.f / (1.f + expf(-x)); }

__global__ void lstm_pointwise(const float* __restrict__ gates,
                               float* __restrict__ h, float* __restrict__ c,
                               _Float16* __restrict__ h16,
                               const int* __restrict__ len, int t,
                               _Float16* __restrict__ outH) {  // outH nullable ([B,LP,H] f16)
  int i = blockIdx.x * blockDim.x + threadIdx.x;
  if (i >= B_ * H_) return;
  int b = i >> 10;
  int j = i & (H_ - 1);
  const float* g = gates + (long)b * G4H_;
  float iv = sigm_(g[j]);
  float fv = sigm_(g[H_ + j]);
  float gv = tanhf(g[2 * H_ + j]);
  float ov = sigm_(g[3 * H_ + j]);
  float cn = fv * c[i] + iv * gv;
  float hn = ov * tanhf(cn);
  bool  m  = t < len[b];
  float hc = m ? hn : h[i];
  float cc = m ? cn : c[i];
  h[i] = hc; c[i] = cc; h16[i] = (_Float16)hc;
  if (outH)
    outH[(long)b * LP_ * H_ + (long)t * H_ + j] = m ? (_Float16)hn : (_Float16)0.f;
}

// ---------------- attention score: a[b,l] = w . tanh(PW[b,l,:] + WH[b,:]) (masked) -----------

__global__ void attn_score(const float* __restrict__ PW, const float* __restrict__ WH,
                           const float* __restrict__ wvec, const int* __restrict__ len,
                           float* __restrict__ a) {
  int r = blockIdx.x * (blockDim.x >> 5) + (threadIdx.x >> 5);
  if (r >= B_ * LP_) return;
  int lane = threadIdx.x & 31;
  int b = r >> 8;
  int l = r & (LP_ - 1);
  float s = 0.f;
  for (int j = lane; j < H_; j += 32)
    s += wvec[j] * tanhf(PW[(long)r * H_ + j] + WH[b * H_ + j]);
#pragma unroll
  for (int m = 16; m; m >>= 1) s += __shfl_xor(s, m, 32);
  if (lane == 0) a[r] = (l < len[b]) ? s : -1e30f;
}

__global__ void softmax256(const float* __restrict__ a, float* __restrict__ alpha) {
  __shared__ float red[LP_];
  int b = blockIdx.x, t = threadIdx.x;
  float v = a[b * LP_ + t];
  red[t] = v; __syncthreads();
  for (int s = 128; s; s >>= 1) { if (t < s) red[t] = fmaxf(red[t], red[t + s]); __syncthreads(); }
  float mx = red[0]; __syncthreads();
  float e = expf(v - mx);
  red[t] = e; __syncthreads();
  for (int s = 128; s; s >>= 1) { if (t < s) red[t] += red[t + s]; __syncthreads(); }
  alpha[b * LP_ + t] = e / red[0];
}

// r[b,j] = sum_l alpha[b,l] * pre_h[b,l,j]
__global__ void weighted_sum(const float* __restrict__ alpha, const _Float16* __restrict__ preH,
                             float* __restrict__ r, _Float16* __restrict__ r16) {
  int i = blockIdx.x * blockDim.x + threadIdx.x;
  if (i >= B_ * H_) return;
  int b = i >> 10;
  int j = i & (H_ - 1);
  const _Float16* p = preH + (long)b * LP_ * H_ + j;
  float s = 0.f;
  for (int l = 0; l < LP_; ++l)
    s += alpha[b * LP_ + l] * (float)p[(long)l * H_];
  r[i] = s; r16[i] = (_Float16)s;
}

// logits = tanh(hs) @ cls_W^T + cls_b ; loss = mean_b (lse - logit[target])
__global__ void logits_loss(const float* __restrict__ hs, const float* __restrict__ clsW,
                            const float* __restrict__ clsb, const int* __restrict__ target,
                            float* __restrict__ out) {
  __shared__ float red[B_];
  int b = threadIdx.x;
  float lg[3];
  for (int cc = 0; cc < 3; ++cc) {
    float s = clsb[cc];
    for (int j = 0; j < H_; ++j) s += tanhf(hs[b * H_ + j]) * clsW[cc * H_ + j];
    lg[cc] = s;
  }
  float mx  = fmaxf(lg[0], fmaxf(lg[1], lg[2]));
  float lse = mx + logf(expf(lg[0] - mx) + expf(lg[1] - mx) + expf(lg[2] - mx));
  red[b] = lse - lg[target[b]];
  __syncthreads();
  for (int s = 32; s; s >>= 1) { if (b < s) red[b] += red[b + s]; __syncthreads(); }
  if (b == 0) out[0] = red[0] / (float)B_;
}

// ---------------- host ----------------

extern "C" void kernel_launch(void* const* d_in, const int* in_sizes, int n_in,
                              void* d_out, int out_size, void* d_ws, size_t ws_size,
                              hipStream_t stream) {
  (void)in_sizes; (void)n_in; (void)out_size; (void)ws_size;
  const int*   pre_x   = (const int*)d_in[0];
  const int*   pre_len = (const int*)d_in[1];
  const int*   hyp_x   = (const int*)d_in[2];
  const int*   hyp_len = (const int*)d_in[3];
  /* d_in[4] hyp2pre == arange(B): identity, unused */
  const int*   target  = (const int*)d_in[5];
  const float* emb     = (const float*)d_in[6];
  const float* A_Wih   = (const float*)d_in[7];
  const float* A_Whh   = (const float*)d_in[8];
  const float* A_b     = (const float*)d_in[9];
  const float* B_Wih   = (const float*)d_in[10];
  const float* B_Whh   = (const float*)d_in[11];
  const float* B_b     = (const float*)d_in[12];
  const float* W_y     = (const float*)d_in[13];
  const float* W_h     = (const float*)d_in[14];
  const float* wvec    = (const float*)d_in[15];
  const float* W_p     = (const float*)d_in[16];
  const float* W_x     = (const float*)d_in[17];
  const float* cls_W   = (const float*)d_in[18];
  const float* cls_b   = (const float*)d_in[19];
  float* out = (float*)d_out;

  char* ws = (char*)d_ws;
  auto carve = [&](size_t bytes) -> char* {
    char* p = ws; ws += (bytes + 255) & ~(size_t)255; return p;
  };
  _Float16* preEmb = (_Float16*)carve((size_t)B_ * LP_ * E_ * 2);
  _Float16* hypEmb = (_Float16*)carve((size_t)B_ * LH_ * E_ * 2);
  _Float16* AWih16 = (_Float16*)carve((size_t)G4H_ * E_ * 2);
  _Float16* AWhh16 = (_Float16*)carve((size_t)G4H_ * H_ * 2);
  _Float16* BWih16 = (_Float16*)carve((size_t)G4H_ * E_ * 2);
  _Float16* BWhh16 = (_Float16*)carve((size_t)G4H_ * H_ * 2);
  _Float16* Wy16   = (_Float16*)carve((size_t)H_ * H_ * 2);
  _Float16* Wh16   = (_Float16*)carve((size_t)H_ * H_ * 2);
  _Float16* Wp16   = (_Float16*)carve((size_t)H_ * H_ * 2);
  _Float16* Wx16   = (_Float16*)carve((size_t)H_ * H_ * 2);
  float*    gates  = (float*)carve((size_t)B_ * G4H_ * 4);
  float*    hbuf   = (float*)carve((size_t)B_ * H_ * 4);
  float*    cbuf   = (float*)carve((size_t)B_ * H_ * 4);
  _Float16* h16    = (_Float16*)carve((size_t)B_ * H_ * 2);
  _Float16* preH16 = (_Float16*)carve((size_t)B_ * LP_ * H_ * 2);
  float*    PW     = (float*)carve((size_t)B_ * LP_ * H_ * 4);
  float*    WH     = (float*)carve((size_t)B_ * H_ * 4);
  float*    abuf   = (float*)carve((size_t)B_ * LP_ * 4);
  float*    alpha  = (float*)carve((size_t)B_ * LP_ * 4);
  float*    rbuf   = (float*)carve((size_t)B_ * H_ * 4);
  _Float16* r16    = (_Float16*)carve((size_t)B_ * H_ * 2);
  float*    hs     = (float*)carve((size_t)B_ * H_ * 4);

  auto castN = [&](const float* src, _Float16* dst, int n) {
    cast_f32_to_f16<<<(n + 255) / 256, 256, 0, stream>>>(src, dst, n);
  };
  castN(A_Wih, AWih16, G4H_ * E_);
  castN(A_Whh, AWhh16, G4H_ * H_);
  castN(B_Wih, BWih16, G4H_ * E_);
  castN(B_Whh, BWhh16, G4H_ * H_);
  castN(W_y, Wy16, H_ * H_);
  castN(W_h, Wh16, H_ * H_);
  castN(W_p, Wp16, H_ * H_);
  castN(W_x, Wx16, H_ * H_);

  {
    int total = B_ * LP_ * E_;
    gather_embed<<<(total + 255) / 256, 256, 0, stream>>>(emb, pre_x, preEmb, total);
    gather_embed<<<(total + 255) / 256, 256, 0, stream>>>(emb, hyp_x, hypEmb, total);
  }
  zero_state<<<(B_ * H_ + 255) / 256, 256, 0, stream>>>(hbuf, cbuf, h16, B_ * H_);

  auto gemm = [&](const _Float16* A, int lda, const _Float16* W, int ldb,
                  float* C, int ldc, int M, int N, int K, const float* bias, int beta) {
    int tiles = (M / 32) * (N / 32);
    wmma_gemm_nt<<<(tiles + 3) / 4, 128, 0, stream>>>(A, lda, W, ldb, C, ldc,
                                                      M / 32, N / 32, K, bias, beta);
  };

  // LSTM A over premise: fused per-step GEMM + pointwise; state in (hbuf,cbuf,h16)
  for (int t = 0; t < LP_; ++t) {
    gemm(preEmb + (size_t)t * E_, LP_ * E_, AWih16, E_, gates, G4H_, B_, G4H_, E_, A_b, 0);
    gemm(h16, H_, AWhh16, H_, gates, G4H_, B_, G4H_, H_, nullptr, 1);
    lstm_pointwise<<<(B_ * H_) / 256, 256, 0, stream>>>(gates, hbuf, cbuf, h16,
                                                        pre_len, t, preH16);
  }
  // LSTM B over hypothesis, continuing from (hA, cA); frozen-state => final h == h_n
  for (int t = 0; t < LH_; ++t) {
    gemm(hypEmb + (size_t)t * E_, LH_ * E_, BWih16, E_, gates, G4H_, B_, G4H_, E_, B_b, 0);
    gemm(h16, H_, BWhh16, H_, gates, G4H_, B_, G4H_, H_, nullptr, 1);
    lstm_pointwise<<<(B_ * H_) / 256, 256, 0, stream>>>(gates, hbuf, cbuf, h16,
                                                        hyp_len, t, nullptr);
  }

  // Attention: PW = pre_h @ W_y^T  (16384x1024x1024), WH = h_n @ W_h^T
  gemm(preH16, H_, Wy16, H_, PW, H_, B_ * LP_, H_, H_, nullptr, 0);
  gemm(h16, H_, Wh16, H_, WH, H_, B_, H_, H_, nullptr, 0);
  attn_score<<<(B_ * LP_) / 8, 256, 0, stream>>>(PW, WH, wvec, pre_len, abuf);
  softmax256<<<B_, LP_, 0, stream>>>(abuf, alpha);
  weighted_sum<<<(B_ * H_) / 256, 256, 0, stream>>>(alpha, preH16, rbuf, r16);

  // h_star pre-activation: hs = r @ W_p^T + h_n @ W_x^T  (tanh fused into loss kernel)
  gemm(r16, H_, Wp16, H_, hs, H_, B_, H_, H_, nullptr, 0);
  gemm(h16, H_, Wx16, H_, hs, H_, B_, H_, H_, nullptr, 1);

  logits_loss<<<1, B_, 0, stream>>>(hs, cls_W, cls_b, target, out);
}
